// HeteroGAT_44006234915466
// MI455X (gfx1250) — matrix operand order repaired and checked by reading, not compiled, not run on hardware
//
#include <hip/hip_runtime.h>
#include <math.h>

#define NNODES 50000
#define CDIM   128
#define HEADS  8
#define EDGES  150000
#define HC     1024   // HEADS*CDIM
#define LLAY   2

typedef __attribute__((ext_vector_type(16))) _Float16 v16h;
typedef __attribute__((ext_vector_type(8)))  _Float16 v8h;
typedef __attribute__((ext_vector_type(8)))  float    v8f;

// sched_group_barrier masks (AMDGPU backend)
#define SG_MFMA  0x008   // MFMA/WMMA group
#define SG_VMEMR 0x020   // VMEM read group

// ---------------------------------------------------------------- utilities
__global__ void fill_f32(float* __restrict__ p, float v, size_t n) {
    size_t i = (size_t)blockIdx.x * blockDim.x + threadIdx.x;
    size_t stride = (size_t)gridDim.x * blockDim.x;
    for (; i < n; i += stride) p[i] = v;
}

__global__ void cvt_f16(const float* __restrict__ x, _Float16* __restrict__ y, size_t n) {
    size_t i = (size_t)blockIdx.x * blockDim.x + threadIdx.x;
    size_t stride = (size_t)gridDim.x * blockDim.x;
    for (; i < n; i += stride) y[i] = (_Float16)x[i];
}

// W [128, M] fp32 (row-major)  ->  Wt [M, 128] f16 (transposed)
__global__ void cvt_wt(const float* __restrict__ W, _Float16* __restrict__ Wt, int M) {
    int idx = blockIdx.x * blockDim.x + threadIdx.x;
    if (idx >= M * CDIM) return;
    int m = idx / CDIM, k = idx - m * CDIM;
    Wt[(size_t)m * CDIM + k] = (_Float16)W[(size_t)k * M + m];
}

// wv[k,h] = sum_c W[k, h*C + c] * a[h, c]       (128x8 outputs)
__global__ void reduce_wv(const float* __restrict__ W, const float* __restrict__ a,
                          float* __restrict__ wv) {
    int idx = blockIdx.x * blockDim.x + threadIdx.x;
    if (idx >= CDIM * HEADS) return;
    int k = idx >> 3, h = idx & 7;
    const float* wr = W + (size_t)k * HC + h * CDIM;
    const float* ar = a + h * CDIM;
    float acc = 0.f;
    for (int c = 0; c < CDIM; ++c) acc += wr[c] * ar[c];
    wv[k * HEADS + h] = acc;
}

// al[n,h] = x[n,:] @ wv[:,h]
__global__ void node_al(const float* __restrict__ x, const float* __restrict__ wv,
                        float* __restrict__ al, int n_nodes) {
    int idx = blockIdx.x * blockDim.x + threadIdx.x;
    if (idx >= n_nodes * HEADS) return;
    int n = idx >> 3, h = idx & 7;
    const float* xr = x + (size_t)n * CDIM;
    float acc = 0.f;
    for (int k = 0; k < CDIM; ++k) acc += xr[k] * wv[k * HEADS + h];
    al[idx] = acc;
}

// --------------------------------------------------- WMMA GEMM (f16 -> f32)
// C[N,M] = A[N,128] * Bt[M,128]^T ; wave computes a 16x64 strip.
// Pipeline shape pinned with sched_group_barrier:
//   [VMEM x16: A(all K) + B(k0)]  then per k-step  [VMEM x8: B(k+1)] [WMMA x4]
__global__ void __launch_bounds__(256, 2)
gemm_wmma_f16(const _Float16* __restrict__ A,
              const _Float16* __restrict__ Bt,
              float* __restrict__ C, int N, int M) {
    const int wave   = threadIdx.x >> 5;
    const int lane   = threadIdx.x & 31;
    const int laneHi = lane >> 4;     // 0 or 1
    const int ml     = lane & 15;
    const int colTiles   = M >> 6;    // 64-wide strips
    const int totalTiles = (N >> 4) * colTiles;
    int tile = blockIdx.x * (blockDim.x >> 5) + wave;   // wave-uniform
    if (tile >= totalTiles) return;
    const int row = (tile / colTiles) << 4;
    const int col = (tile % colTiles) << 6;

    // Preload all A fragments: K contiguous [laneHi*16,+16) per 32-K step
    const _Float16* arow = A + (size_t)(row + ml) * CDIM;
    v16h af[4];
    #pragma unroll
    for (int ks = 0; ks < 4; ++ks) {
        v8h a0 = *(const v8h*)(arow + ks * 32 + laneHi * 16);
        v8h a1 = *(const v8h*)(arow + ks * 32 + laneHi * 16 + 8);
        #pragma unroll
        for (int i = 0; i < 8; ++i) { af[ks][i] = a0[i]; af[ks][8 + i] = a1[i]; }
    }

    // B fragment base pointers (column N = lane, K contiguous from laneHi*16)
    const _Float16* bbase[4];
    #pragma unroll
    for (int t = 0; t < 4; ++t)
        bbase[t] = Bt + (size_t)(col + t * 16 + ml) * CDIM + laneHi * 16;

    auto loadB = [&](v16h* bf, int kb) {
        #pragma unroll
        for (int t = 0; t < 4; ++t) {
            v8h b0 = *(const v8h*)(bbase[t] + kb);
            v8h b1 = *(const v8h*)(bbase[t] + kb + 8);
            #pragma unroll
            for (int i = 0; i < 8; ++i) { bf[t][i] = b0[i]; bf[t][8 + i] = b1[i]; }
        }
    };

    v8f acc[4] = {};
    v16h bcur[4], bnext[4];
    loadB(bcur, 0);
    // group 0: the 16 preload VMEM reads (A fragments + B k-step 0)
    __builtin_amdgcn_sched_group_barrier(SG_VMEMR, 16, 0);
    #pragma unroll
    for (int ks = 0; ks < 4; ++ks) {
        if (ks < 3) {
            loadB(bnext, (ks + 1) * 32);   // prefetch next k-step
            __builtin_amdgcn_sched_group_barrier(SG_VMEMR, 8, 0);
        }
        #pragma unroll
        for (int t = 0; t < 4; ++t) {
            acc[t] = __builtin_amdgcn_wmma_f32_16x16x32_f16(
                false, af[ks], false, bcur[t], (short)0, acc[t], false, false);
        }
        __builtin_amdgcn_sched_group_barrier(SG_MFMA, 4, 0);
        #pragma unroll
        for (int t = 0; t < 4; ++t) bcur[t] = bnext[t];
    }
    #pragma unroll
    for (int t = 0; t < 4; ++t) {
        #pragma unroll
        for (int v = 0; v < 8; ++v) {
            int r  = row + v + laneHi * 8;
            int cc = col + t * 16 + ml;
            C[(size_t)r * M + cc] = acc[t][v];
        }
    }
}

// ------------------------------------------------------------- edge kernels
__device__ __forceinline__ void atomicMaxFloat(float* addr, float v) {
    if (v >= 0.0f) atomicMax((int*)addr, __float_as_int(v));
    else           atomicMin((unsigned int*)addr, __float_as_uint(v));
}

__global__ void edge_logits_max(const int* __restrict__ ei,
                                const float* __restrict__ als, const float* __restrict__ ald,
                                float* __restrict__ ebuf, float* __restrict__ maxb) {
    int idx = blockIdx.x * blockDim.x + threadIdx.x;
    if (idx >= EDGES * HEADS) return;
    int e = idx >> 3, h = idx & 7;
    int src = ei[e], dst = ei[EDGES + e];
    float v = als[src * HEADS + h] + ald[dst * HEADS + h];
    v = v > 0.0f ? v : 0.2f * v;                       // leaky_relu(0.2)
    ebuf[idx] = v;
    atomicMaxFloat(&maxb[dst * HEADS + h], v);
}

__global__ void edge_exp_sum(const int* __restrict__ ei, float* __restrict__ ebuf,
                             const float* __restrict__ maxb, float* __restrict__ sumb) {
    int idx = blockIdx.x * blockDim.x + threadIdx.x;
    if (idx >= EDGES * HEADS) return;
    int e = idx >> 3, h = idx & 7;
    int dst = ei[EDGES + e];
    float w = expf(ebuf[idx] - maxb[dst * HEADS + h]);
    ebuf[idx] = w;
    atomicAdd(&sumb[dst * HEADS + h], w);
}

// one 128-thread block per edge; accumulates head-mean message directly
__global__ void edge_scatter(const int* __restrict__ ei, const float* __restrict__ ebuf,
                             const float* __restrict__ sumb, const float* __restrict__ hs,
                             float* __restrict__ obuf) {
    int e = blockIdx.x;
    int c = threadIdx.x;
    __shared__ float alpha[HEADS];
    int dst = ei[EDGES + e];
    if (c < HEADS)
        alpha[c] = ebuf[e * HEADS + c] / (sumb[dst * HEADS + c] + 1e-16f) * (1.0f / HEADS);
    __syncthreads();
    int src = ei[e];
    const float* hrow = hs + (size_t)src * HC + c;
    float acc = 0.f;
    #pragma unroll
    for (int h = 0; h < HEADS; ++h) acc += alpha[h] * hrow[h * CDIM];
    atomicAdd(&obuf[(size_t)dst * CDIM + c], acc);
}

// --------------------------------------------------- residual + LN + ELU + skip
__global__ void finalize_node(const float* __restrict__ obuf, const float* __restrict__ rbuf,
                              const float* __restrict__ bias, const float* __restrict__ g,
                              const float* __restrict__ b, const float* __restrict__ xold,
                              float* __restrict__ xnew) {
    int n = blockIdx.x, c = threadIdx.x;
    __shared__ float s1[CDIM], s2[CDIM];
    float t = obuf[(size_t)n * CDIM + c] + bias[c] + rbuf[(size_t)n * CDIM + c];
    s1[c] = t; s2[c] = t * t;
    __syncthreads();
    for (int off = 64; off > 0; off >>= 1) {
        if (c < off) { s1[c] += s1[c + off]; s2[c] += s2[c + off]; }
        __syncthreads();
    }
    float mean = s1[0] * (1.0f / CDIM);
    float var  = s2[0] * (1.0f / CDIM) - mean * mean;
    float y = (t - mean) * rsqrtf(var + 1e-5f) * g[c] + b[c];
    y = y > 0.f ? y : (expf(y) - 1.f);                 // elu
    xnew[(size_t)n * CDIM + c] = y + xold[(size_t)n * CDIM + c];
}

// -------------------------------------------------------------------- host
extern "C" void kernel_launch(void* const* d_in, const int* in_sizes, int n_in,
                              void* d_out, int out_size, void* d_ws, size_t ws_size,
                              hipStream_t stream) {
    const float* x_u0    = (const float*)d_in[0];
    const float* x_i0    = (const float*)d_in[1];
    const int*   ei_ui   = (const int*)d_in[2];
    const int*   ei_iu   = (const int*)d_in[3];
    const float* Wsrc_ui = (const float*)d_in[4];
    const float* Wdst_ui = (const float*)d_in[5];
    const float* asrc_ui = (const float*)d_in[6];
    const float* adst_ui = (const float*)d_in[7];
    const float* b_ui    = (const float*)d_in[8];
    const float* Wsrc_iu = (const float*)d_in[9];
    const float* Wdst_iu = (const float*)d_in[10];
    const float* asrc_iu = (const float*)d_in[11];
    const float* adst_iu = (const float*)d_in[12];
    const float* b_iu    = (const float*)d_in[13];
    const float* Ru      = (const float*)d_in[14];
    const float* Ri      = (const float*)d_in[15];
    const float* gu      = (const float*)d_in[16];
    const float* bu      = (const float*)d_in[17];
    const float* gi      = (const float*)d_in[18];
    const float* bi      = (const float*)d_in[19];

    char* ws = (char*)d_ws; size_t off = 0;
    auto alloc = [&](size_t bytes) -> void* {
        void* p = ws + off; off += (bytes + 255) & ~(size_t)255; return p;
    };
    float*     hs   = (float*)alloc((size_t)NNODES * HC * 4);
    _Float16*  xh_u = (_Float16*)alloc((size_t)NNODES * CDIM * 2);
    _Float16*  xh_i = (_Float16*)alloc((size_t)NNODES * CDIM * 2);
    _Float16*  wt   = (_Float16*)alloc((size_t)HC * CDIM * 2);
    _Float16*  rt   = (_Float16*)alloc((size_t)CDIM * CDIM * 2);
    float*     wv_s = (float*)alloc(CDIM * HEADS * 4);
    float*     wv_d = (float*)alloc(CDIM * HEADS * 4);
    float*     al_s = (float*)alloc((size_t)NNODES * HEADS * 4);
    float*     al_d = (float*)alloc((size_t)NNODES * HEADS * 4);
    float*     ebuf = (float*)alloc((size_t)EDGES * HEADS * 4);
    float*     maxb = (float*)alloc((size_t)NNODES * HEADS * 4);
    float*     sumb = (float*)alloc((size_t)NNODES * HEADS * 4);
    float*     o_u  = (float*)alloc((size_t)NNODES * CDIM * 4);
    float*     o_i  = (float*)alloc((size_t)NNODES * CDIM * 4);
    float*     rbuf = (float*)alloc((size_t)NNODES * CDIM * 4);
    float*     xu1  = (float*)alloc((size_t)NNODES * CDIM * 4);
    float*     xi1  = (float*)alloc((size_t)NNODES * CDIM * 4);

    const int BT = 256;
    auto cdiv = [](size_t a, size_t b) { return (int)((a + b - 1) / b); };
    const size_t NC = (size_t)NNODES * CDIM;
    const size_t NH = (size_t)NNODES * HEADS;
    const size_t EH = (size_t)EDGES * HEADS;
    const float NEG_INF = -__builtin_huge_valf();

    // one GAT: src features -> dst aggregation (writes obuf)
    auto gat = [&](const _Float16* xh_src, const float* x_src, const float* x_dst,
                   const int* ei, const float* Wsrc, const float* Wdst,
                   const float* asrc, const float* adst, float* obuf, int l) {
        const float* Wsrc_l = Wsrc + (size_t)l * CDIM * HC;
        const float* Wdst_l = Wdst + (size_t)l * CDIM * HC;
        cvt_wt<<<cdiv((size_t)HC * CDIM, BT), BT, 0, stream>>>(Wsrc_l, wt, HC);
        reduce_wv<<<cdiv(CDIM * HEADS, BT), BT, 0, stream>>>(Wsrc_l, asrc + (size_t)l * HEADS * CDIM, wv_s);
        reduce_wv<<<cdiv(CDIM * HEADS, BT), BT, 0, stream>>>(Wdst_l, adst + (size_t)l * HEADS * CDIM, wv_d);
        node_al<<<cdiv(NH, BT), BT, 0, stream>>>(x_src, wv_s, al_s, NNODES);
        node_al<<<cdiv(NH, BT), BT, 0, stream>>>(x_dst, wv_d, al_d, NNODES);
        // hs = x_src @ Wsrc   [50000,1024]
        int tiles = (NNODES / 16) * (HC / 64);
        gemm_wmma_f16<<<cdiv(tiles, 8), 256, 0, stream>>>(xh_src, wt, hs, NNODES, HC);
        fill_f32<<<cdiv(NH, BT), BT, 0, stream>>>(maxb, NEG_INF, NH);
        fill_f32<<<cdiv(NH, BT), BT, 0, stream>>>(sumb, 0.f, NH);
        fill_f32<<<cdiv(NC, BT), BT, 0, stream>>>(obuf, 0.f, NC);
        edge_logits_max<<<cdiv(EH, BT), BT, 0, stream>>>(ei, al_s, al_d, ebuf, maxb);
        edge_exp_sum<<<cdiv(EH, BT), BT, 0, stream>>>(ei, ebuf, maxb, sumb);
        edge_scatter<<<EDGES, CDIM, 0, stream>>>(ei, ebuf, sumb, hs, obuf);
    };

    for (int l = 0; l < LLAY; ++l) {
        const float* xu = l ? xu1 : x_u0;
        const float* xi = l ? xi1 : x_i0;
        float* xu_next = (l == LLAY - 1) ? (float*)d_out : xu1;
        float* xi_next = (l == LLAY - 1) ? ((float*)d_out + NC) : xi1;

        cvt_f16<<<cdiv(NC, BT), BT, 0, stream>>>(xu, xh_u, NC);
        cvt_f16<<<cdiv(NC, BT), BT, 0, stream>>>(xi, xh_i, NC);

        // o_i = GAT_ui(x_u -> x_i),  o_u = GAT_iu(x_i -> x_u)  (both on old x)
        gat(xh_u, xu, xi, ei_ui, Wsrc_ui, Wdst_ui, asrc_ui, adst_ui, o_i, l);
        gat(xh_i, xi, xu, ei_iu, Wsrc_iu, Wdst_iu, asrc_iu, adst_iu, o_u, l);

        int rtiles = (NNODES / 16) * (CDIM / 64);
        // user update: y_u = elu(LN(o_u + x_u@Ru)) + x_u
        cvt_wt<<<cdiv((size_t)CDIM * CDIM, BT), BT, 0, stream>>>(Ru + (size_t)l * CDIM * CDIM, rt, CDIM);
        gemm_wmma_f16<<<cdiv(rtiles, 8), 256, 0, stream>>>(xh_u, rt, rbuf, NNODES, CDIM);
        finalize_node<<<NNODES, CDIM, 0, stream>>>(o_u, rbuf, b_iu + (size_t)l * CDIM,
                                                   gu + (size_t)l * CDIM, bu + (size_t)l * CDIM,
                                                   xu, xu_next);
        // item update: y_i = elu(LN(o_i + x_i@Ri)) + x_i
        cvt_wt<<<cdiv((size_t)CDIM * CDIM, BT), BT, 0, stream>>>(Ri + (size_t)l * CDIM * CDIM, rt, CDIM);
        gemm_wmma_f16<<<cdiv(rtiles, 8), 256, 0, stream>>>(xh_i, rt, rbuf, NNODES, CDIM);
        finalize_node<<<NNODES, CDIM, 0, stream>>>(o_i, rbuf, b_ui + (size_t)l * CDIM,
                                                   gi + (size_t)l * CDIM, bi + (size_t)l * CDIM,
                                                   xi, xi_next);
    }
    (void)in_sizes; (void)n_in; (void)out_size; (void)ws_size;
}
